// Discriminator_3367254360439
// MI455X (gfx1250) — matrix operand order repaired and checked by reading, not compiled, not run on hardware
//
#include <hip/hip_runtime.h>
#include <hip/hip_bf16.h>

// ---------------------------------------------------------------------------
// CDNA5 (gfx1250) graph-conv discriminator.
// Convs = implicit GEMM on v_wmma_f32_16x16x32_bf16 with BOTH operands
// prepacked into the ISA fragment lane layouts (CDNA5 ISA 7.12.2), so the
// hot kernel is pure b128 loads + WMMA:
//   A: pack_w_kernel    -> Aw[cg][s][lane][16]  (weights, bf16)
//   B: im2col_pack_knl  -> Bp[img][tile][s][lane][16] (pad+cvt+im2col fused)
//   GEMM: per k-step: 2 b128 (A) + 4x2 b128 (B) + 4 WMMA, no branches.
// ---------------------------------------------------------------------------

typedef __attribute__((ext_vector_type(16))) __bf16 v16bf;
typedef __attribute__((ext_vector_type(8)))  float  v8f;

union Frag16 {
    v16bf v;
    unsigned short u[16];
    uint4 q[2];
};

__device__ __forceinline__ unsigned short f2bf(float f) {
    unsigned int u = __builtin_bit_cast(unsigned int, f);
    unsigned int r = u + 0x7FFFu + ((u >> 16) & 1u);   // RNE bf16
    return (unsigned short)(r >> 16);
}

// ---------------------------------------------------------------------------
// Weight prepack: A fragment layout (16-bit A 16x32).
//  lane<16: K={0..7,16..23} of row M=lane ; lane>=16: K={8..15,24..31}
// ---------------------------------------------------------------------------
__global__ void pack_w_kernel(const float* __restrict__ wgt, // [Cout,Cin,3,3]
                              unsigned short* __restrict__ Aw,
                              int Cin, int ksteps, int tot)
{
    const int idx = blockIdx.x * blockDim.x + threadIdx.x;
    if (idx >= tot) return;
    const int j    = idx & 15;
    const int lane = (idx >> 4) & 31;
    const int s    = (idx >> 9) % ksteps;
    const int cg   = idx / (ksteps * 512);
    const int m    = cg * 16 + (lane & 15);
    const int k    = s * 32 + (lane >> 4) * 8 + j + ((j >= 8) ? 8 : 0);
    const int K    = Cin * 9;
    float val = 0.0f;
    if (k < K) {
        const int ci = k / 9;
        val = wgt[(m * Cin + ci) * 9 + (k - ci * 9)];
    }
    Aw[idx] = f2bf(val);
}

// ---------------------------------------------------------------------------
// Fused pad + f32->bf16 + im2col into B fragment layout (16-bit B 32x16).
//  frag elem j of lane = B[(lane>=16 ? 8:0) + j + (j>=8 ? 8:0)][tile*16 + (lane&15)]
// ---------------------------------------------------------------------------
__global__ void im2col_pack_kernel(const float* __restrict__ in, // [64,Cin,H,H]
                                   unsigned short* __restrict__ Bp,
                                   int Cin, int H, int stride,
                                   int ksteps, int ntiles, int tot)
{
    const int idx = blockIdx.x * blockDim.x + threadIdx.x;
    if (idx >= tot) return;
    const int j    = idx & 15;
    const int lane = (idx >> 4) & 31;
    int r = idx >> 9;                 // (img*ntiles + tile)*ksteps + s
    const int s    = r % ksteps; r /= ksteps;
    const int tile = r % ntiles;
    const int img  = r / ntiles;

    const int k = s * 32 + (lane >> 4) * 8 + j + ((j >= 8) ? 8 : 0);
    const int Ho = (stride == 2) ? (H >> 1) : H;
    const int npix = Ho * Ho;
    const int p = tile * 16 + (lane & 15);
    const int K = Cin * 9;

    float val = 0.0f;
    if (k < K && p < npix) {
        const int ci = k / 9;
        const int rs = k - ci * 9;
        const int ry = rs / 3;
        const int rx = rs - ry * 3;
        const int oy = p / Ho;
        const int ox = p - oy * Ho;
        const int iy = oy * stride + ry - 1;
        const int ix = ox * stride + rx - 1;
        if (iy >= 0 && iy < H && ix >= 0 && ix < H)
            val = in[((img * Cin + ci) * H + iy) * H + ix];
    }
    Bp[idx] = f2bf(val);
}

// ---------------------------------------------------------------------------
// Pure-GEMM conv: one wave32 = 16 cout x 64 pixels (4 tiles, 4 accumulators).
// ---------------------------------------------------------------------------
__global__ __launch_bounds__(32)
void gemm_conv_wmma(const unsigned short* __restrict__ Aw,
                    const unsigned short* __restrict__ Bp,
                    const float* __restrict__ bias,
                    float* __restrict__ out,              // [64,Cout,Ho,Ho]
                    int Cout, int Ho, int ksteps, int ntiles)
{
    const int lane = threadIdx.x;
    const int hi = lane >> 4, lo = lane & 15;
    const int img = blockIdx.z, cg = blockIdx.y;
    const int npix = Ho * Ho;

    const uint4* ap = (const uint4*)(Aw + (size_t)cg * ksteps * 512 + lane * 16);

    const int tile0 = blockIdx.x * 4;
    const int tc0 = (tile0 + 0 < ntiles) ? tile0 + 0 : ntiles - 1;
    const int tc1 = (tile0 + 1 < ntiles) ? tile0 + 1 : ntiles - 1;
    const int tc2 = (tile0 + 2 < ntiles) ? tile0 + 2 : ntiles - 1;
    const int tc3 = (tile0 + 3 < ntiles) ? tile0 + 3 : ntiles - 1;
    const uint4* bp0 = (const uint4*)(Bp + ((size_t)img * ntiles + tc0) * ksteps * 512 + lane * 16);
    const uint4* bp1 = (const uint4*)(Bp + ((size_t)img * ntiles + tc1) * ksteps * 512 + lane * 16);
    const uint4* bp2 = (const uint4*)(Bp + ((size_t)img * ntiles + tc2) * ksteps * 512 + lane * 16);
    const uint4* bp3 = (const uint4*)(Bp + ((size_t)img * ntiles + tc3) * ksteps * 512 + lane * 16);

    v8f acc0 = {}, acc1 = {}, acc2 = {}, acc3 = {};

#pragma unroll 2
    for (int s = 0; s < ksteps; ++s) {
        Frag16 a, b0, b1, b2, b3;
        a.q[0]  = ap[0];  a.q[1]  = ap[1];  ap  += 32;   // 512 u16 per step
        b0.q[0] = bp0[0]; b0.q[1] = bp0[1]; bp0 += 32;
        b1.q[0] = bp1[0]; b1.q[1] = bp1[1]; bp1 += 32;
        b2.q[0] = bp2[0]; b2.q[1] = bp2[1]; bp2 += 32;
        b3.q[0] = bp3[0]; b3.q[1] = bp3[1]; bp3 += 32;
        __builtin_prefetch(ap + 32, 0, 1);               // next A block
        acc0 = __builtin_amdgcn_wmma_f32_16x16x32_bf16(false, a.v, false, b0.v, (short)0, acc0, false, false);
        acc1 = __builtin_amdgcn_wmma_f32_16x16x32_bf16(false, a.v, false, b1.v, (short)0, acc1, false, false);
        acc2 = __builtin_amdgcn_wmma_f32_16x16x32_bf16(false, a.v, false, b2.v, (short)0, acc2, false, false);
        acc3 = __builtin_amdgcn_wmma_f32_16x16x32_bf16(false, a.v, false, b3.v, (short)0, acc3, false, false);
    }

    // epilogue: bias + leaky ReLU; store index is simply orow + p (row-major)
    const int p0 = (tile0 + 0) * 16 + lo;
    const int p1 = (tile0 + 1) * 16 + lo;
    const int p2 = (tile0 + 2) * 16 + lo;
    const int p3 = (tile0 + 3) * 16 + lo;
#pragma unroll
    for (int r = 0; r < 8; ++r) {
        const int mo = cg * 16 + r + hi * 8;
        const float bv = bias[mo];
        const size_t orow = ((size_t)img * Cout + mo) * npix;
        float v;
        v = acc0[r] + bv; v = (v > 0.0f) ? v : 0.1f * v;
        if (p0 < npix) out[orow + p0] = v;
        v = acc1[r] + bv; v = (v > 0.0f) ? v : 0.1f * v;
        if (p1 < npix) out[orow + p1] = v;
        v = acc2[r] + bv; v = (v > 0.0f) ? v : 0.1f * v;
        if (p2 < npix) out[orow + p2] = v;
        v = acc3[r] + bv; v = (v > 0.0f) ? v : 0.1f * v;
        if (p3 < npix) out[orow + p3] = v;
    }
}

// ---------------------------------------------------------------------------
// y = given_y @ l1_w^T + l1_b -> channels 1..8 of concat input
// ---------------------------------------------------------------------------
__global__ void linear_y_kernel(const float* __restrict__ gy,
                                const float* __restrict__ w,
                                const float* __restrict__ b,
                                float* __restrict__ xcat)
{
    const int idx = blockIdx.x * blockDim.x + threadIdx.x;
    if (idx >= 64 * 32768) return;
    const int v = idx >> 15;
    const int o = idx & 32767;
    float s = b[o];
#pragma unroll
    for (int k = 0; k < 18; ++k) s += gy[v * 18 + k] * w[o * 18 + k];
    xcat[(size_t)v * 9 * 4096 + 4096 + o] = s;
}

__global__ void copy_x_kernel(const float* __restrict__ x, float* __restrict__ xcat)
{
    const int idx = blockIdx.x * blockDim.x + threadIdx.x;
    if (idx >= 64 * 4096) return;
    xcat[(size_t)(idx >> 12) * 9 * 4096 + (idx & 4095)] = x[idx];
}

// ---------------------------------------------------------------------------
// CMP message passing (signed symmetric edges, float atomics)
// ---------------------------------------------------------------------------
__global__ void init_h_kernel(const float* __restrict__ feats,
                              float* __restrict__ h, int C, int HW)
{
    const int idx = blockIdx.x * blockDim.x + threadIdx.x;
    const int per = 3 * C * HW;
    if (idx >= 64 * per) return;
    const int v = idx / per;
    const int r = idx - v * per;
    const int c3 = r / HW;
    h[idx] = (c3 < C) ? feats[(v * C + c3) * HW + (r - c3 * HW)] : 0.0f;
}

__global__ void cmp_scatter_kernel(const float* __restrict__ feats,
                                   const int* __restrict__ edges,
                                   float* __restrict__ h, int C, int HW)
{
    const int idx = blockIdx.x * blockDim.x + threadIdx.x;
    const int per = C * HW;
    if (idx >= 2 * 256 * per) return;
    const int e   = idx / per;
    const int off = idx - e * per;
    const int eid = e & 255;
    const int dir = e >> 8;
    const int a  = edges[eid * 3 + 0];
    const int sg = edges[eid * 3 + 1];
    const int d  = edges[eid * 3 + 2];
    const int src = dir ? d : a;
    const int dst = dir ? a : d;
    const int band = (sg > 0) ? C : 2 * C;
    atomicAdd(&h[((size_t)dst * 3 * C + band) * HW + off],
              feats[(size_t)src * C * HW + off]);
}

// ---------------------------------------------------------------------------
// Head: [256,128] -> out[8]
// ---------------------------------------------------------------------------
__global__ __launch_bounds__(256)
void head_kernel(const float* __restrict__ xr, const int* __restrict__ nd,
                 const float* __restrict__ fcg_w, const float* __restrict__ fcg_b,
                 const float* __restrict__ fcl_w, const float* __restrict__ fcl_b,
                 float* __restrict__ out)
{
    __shared__ float og[8];
    const int t = threadIdx.x;
    if (t < 8) og[t] = 0.0f;
    __syncthreads();
    float dg = 0.0f, dl = 0.0f;
    for (int k = 0; k < 128; ++k) {
        const float v = xr[t * 128 + k];
        dg += v * fcg_w[k];
        dl += v * fcl_w[k];
    }
    atomicAdd(&og[nd[t]], dg + 5.0f * (dl + fcl_b[0]));
    __syncthreads();
    if (t < 8) out[t] = og[t] + fcg_b[0];
}

// ---------------------------------------------------------------------------
// Host orchestration
// ---------------------------------------------------------------------------
struct Ctx {
    unsigned short* Aw;
    unsigned short* Bp;
    hipStream_t stream;
};

static void conv_layer(const Ctx& c, const float* in, const float* w,
                       const float* b, float* out, int Cin, int H, int Cout,
                       int stride)
{
    const int K = Cin * 9;
    const int ksteps = (K + 31) / 32;
    const int Ho = (stride == 2) ? (H >> 1) : H;
    const int npix = Ho * Ho;
    const int ntiles = (npix + 15) / 16;

    const int tot_a = (Cout / 16) * ksteps * 512;
    pack_w_kernel<<<(tot_a + 255) / 256, 256, 0, c.stream>>>(w, c.Aw, Cin, ksteps, tot_a);

    const int tot_b = 64 * ntiles * ksteps * 512;
    im2col_pack_kernel<<<(tot_b + 255) / 256, 256, 0, c.stream>>>(
        in, c.Bp, Cin, H, stride, ksteps, ntiles, tot_b);

    dim3 grid((ntiles + 3) / 4, Cout / 16, 64);
    gemm_conv_wmma<<<grid, 32, 0, c.stream>>>(c.Aw, c.Bp, b, out, Cout, Ho,
                                              ksteps, ntiles);
}

static void cmp_block(const float* feats, const int* edges, float* h, int C,
                      int HW, hipStream_t stream)
{
    const int tot_i = 64 * 3 * C * HW;
    init_h_kernel<<<(tot_i + 255) / 256, 256, 0, stream>>>(feats, h, C, HW);
    const int tot_s = 2 * 256 * C * HW;
    cmp_scatter_kernel<<<(tot_s + 255) / 256, 256, 0, stream>>>(feats, edges, h, C, HW);
}

extern "C" void kernel_launch(void* const* d_in, const int* in_sizes, int n_in,
                              void* d_out, int out_size, void* d_ws, size_t ws_size,
                              hipStream_t stream)
{
    const float* x       = (const float*)d_in[0];
    const float* given_y = (const float*)d_in[1];
    const int*   given_w = (const int*)  d_in[2];
    const int*   nd      = (const int*)  d_in[3];
    const float* l1_w    = (const float*)d_in[4];
    const float* l1_b    = (const float*)d_in[5];
    const float* fcg_w = (const float*)d_in[44];
    const float* fcg_b = (const float*)d_in[45];
    const float* fcl_w = (const float*)d_in[46];
    const float* fcl_b = (const float*)d_in[47];
    auto W = [&](int i) { return (const float*)d_in[i]; };

    // workspace (floats): buf0 | buf1 | Aw(512K u16) | Bp(118M u16)
    const size_t NB = (size_t)64 * 48 * 64 * 64;    // 12.58M floats each
    float* buf0 = (float*)d_ws;
    float* buf1 = buf0 + NB;
    Ctx c;
    c.Aw = (unsigned short*)(buf1 + NB);
    c.Bp = c.Aw + (size_t)512 * 1024;
    c.stream = stream;

    // input concat [64,9,64,64] -> buf0
    copy_x_kernel<<<(64 * 4096 + 255) / 256, 256, 0, stream>>>(x, buf0);
    linear_y_kernel<<<(64 * 32768 + 255) / 256, 256, 0, stream>>>(given_y, l1_w, l1_b, buf0);

    // encoder @64
    conv_layer(c, buf0, W(6),  W(7),  buf1,  9, 64, 16, 1);
    conv_layer(c, buf1, W(8),  W(9),  buf0, 16, 64, 16, 1);
    conv_layer(c, buf0, W(10), W(11), buf1, 16, 64, 16, 1);
    conv_layer(c, buf1, W(12), W(13), buf0, 16, 64, 16, 1);     // feats: buf0

    // cmp1 @64
    cmp_block(buf0, given_w, buf1, 16, 4096, stream);
    conv_layer(c, buf1, W(14), W(15), buf0, 48, 64, 32, 1);
    conv_layer(c, buf0, W(16), W(17), buf1, 32, 64, 32, 1);
    conv_layer(c, buf1, W(18), W(19), buf0, 32, 64, 16, 1);
    conv_layer(c, buf0, W(20), W(21), buf1, 16, 64, 16, 2);     // -> 32x32, buf1

    // cmp2 @32
    cmp_block(buf1, given_w, buf0, 16, 1024, stream);
    conv_layer(c, buf0, W(22), W(23), buf1, 48, 32, 32, 1);
    conv_layer(c, buf1, W(24), W(25), buf0, 32, 32, 32, 1);
    conv_layer(c, buf0, W(26), W(27), buf1, 32, 32, 16, 1);
    conv_layer(c, buf1, W(28), W(29), buf0, 16, 32, 16, 2);     // -> 16x16, buf0

    // cmp3 @16
    cmp_block(buf0, given_w, buf1, 16, 256, stream);
    conv_layer(c, buf1, W(30), W(31), buf0, 48, 16, 32, 1);
    conv_layer(c, buf0, W(32), W(33), buf1, 32, 16, 32, 1);
    conv_layer(c, buf1, W(34), W(35), buf0, 32, 16, 16, 1);
    conv_layer(c, buf0, W(36), W(37), buf1, 16, 16, 16, 2);     // -> 8x8, buf1

    // decoder
    conv_layer(c, buf1, W(38), W(39), buf0,  16, 8, 256, 2);    // [64,256,4,4]
    conv_layer(c, buf0, W(40), W(41), buf1, 256, 4, 128, 2);    // [64,128,2,2]
    conv_layer(c, buf1, W(42), W(43), buf0, 128, 2, 128, 1);    // [256,128]

    head_kernel<<<1, 256, 0, stream>>>(buf0, nd, fcg_w, fcg_b, fcl_w, fcl_b,
                                       (float*)d_out);
}
